// ScaledDotProductAttention_12068858102083
// MI455X (gfx1250) — compile-verified, matrix operand order checked by reading, not run on hardware
//
#include <hip/hip_runtime.h>
#include <hip/hip_bf16.h>

typedef __attribute__((ext_vector_type(16))) _Float16 v16h;
typedef __attribute__((ext_vector_type(8)))  _Float16 v8h;
typedef __attribute__((ext_vector_type(8)))  float    v8f;
typedef __attribute__((ext_vector_type(4)))  float    v4f;

#define B_ 64
#define L_ 2048
#define D_ 128
#define SSTRIDE 2052      // 2048 + 4: r*2052 % 64 = 4r -> spreads banks across rows
#define VT_STRIDE 40      // 32 rows + 8 pad halves: col*40*2B = 80B*c -> 16B aligned
#define VT_BUF (128 * VT_STRIDE)

// one workgroup = one (batch, 16-row q tile); 4 waves of 32 (wave32)

__device__ __forceinline__ void stage_v_tile(_Float16* __restrict__ dst,
                                             const float* __restrict__ vb,
                                             int kk0, int tid)
{
    // 32x128 f32 tile -> transposed f16 in LDS: dst[col*VT_STRIDE + row]
    const int rr  = tid & 31;    // row within tile
    const int seg = tid >> 5;    // column block 0..3 (32 cols each)
    const float* src = vb + (size_t)(kk0 + rr) * D_ + seg * 32;
    _Float16* d = dst + (seg * 32) * VT_STRIDE + rr;
    #pragma unroll
    for (int j = 0; j < 32; j += 4) {
        v4f x = *(const v4f*)(src + j);
        d[(j + 0) * VT_STRIDE] = (_Float16)x[0];
        d[(j + 1) * VT_STRIDE] = (_Float16)x[1];
        d[(j + 2) * VT_STRIDE] = (_Float16)x[2];
        d[(j + 3) * VT_STRIDE] = (_Float16)x[3];
    }
}

__global__ __launch_bounds__(128)
void attn_fused_kernel(const float* __restrict__ Q,
                       const float* __restrict__ K,
                       const float* __restrict__ V,
                       const unsigned char* __restrict__ MASK,
                       float* __restrict__ OUT,
                       float* __restrict__ ATTN)
{
    extern __shared__ float smem[];
    float*     Sp     = smem;                    // [16][SSTRIDE] score / P panel
    float*     red    = smem + 16 * SSTRIDE;     // [16][8]
    float*     rowmax = red + 128;               // [16]
    float*     rowinv = rowmax + 16;             // [16]
    _Float16*  Vt     = (_Float16*)(rowinv + 16);// 2 x [128][VT_STRIDE] f16 (double buf)

    const int tid = threadIdx.x;
    const int ln  = tid & 31;
    const int wv  = tid >> 5;
    const int wg  = blockIdx.x;
    const int b   = wg >> 7;            // 128 q tiles per batch
    const int q0  = (wg & 127) * 16;

    const int m  = ln & 15;             // row (A) or col (B/C) index for this lane
    const int hi = ln >> 4;
    const int lb = hi ? 8 : 0;

    const float* qb = Q + (size_t)b * L_ * D_;
    const float* kb = K + (size_t)b * L_ * D_;
    const float* vb = V + (size_t)b * L_ * D_;

    // ---------------- Phase 0: Q tile as f16 A-fragments ----------------
    v16h afrag[4];
    {
        const float* qrow = qb + (size_t)(q0 + m) * D_;
        #pragma unroll
        for (int c = 0; c < 4; ++c) {
            const int base = 32 * c + lb;
            v16h t;
            #pragma unroll
            for (int j = 0; j < 8; ++j) {
                t[j]     = (_Float16)qrow[base + j];
                t[8 + j] = (_Float16)qrow[base + 16 + j];
            }
            afrag[c] = t;
        }
    }

    const float scale = 0.08838834764831845f; // 1/sqrt(128)

    // ---------------- Phase 1: S = (QK^T)/sqrt(D) into LDS panel ----------
    for (int kt = wv; kt < L_ / 16; kt += 4) {
        const int n0 = kt * 16;
        v8f acc = {};
        const float* krow = kb + (size_t)(n0 + m) * D_; // lane m = column n
        #pragma unroll
        for (int c = 0; c < 4; ++c) {
            const int base = 32 * c + lb;
            v16h bfrag;
            #pragma unroll
            for (int j = 0; j < 8; ++j) {
                bfrag[j]     = (_Float16)krow[base + j];
                bfrag[8 + j] = (_Float16)krow[base + 16 + j];
            }
            acc = __builtin_amdgcn_wmma_f32_16x16x32_f16(
                      false, afrag[c], false, bfrag, (short)0, acc, false, false);
        }
        #pragma unroll
        for (int vv = 0; vv < 8; ++vv)
            Sp[(hi * 8 + vv) * SSTRIDE + n0 + m] = acc[vv] * scale;
    }
    __syncthreads();

    // ------- Phase 2: apply mask (coalesced), then row max / sum-exp -------
    {
        const int r = tid >> 3;   // 8 threads per row, interleaved float4 cols
        const int p = tid & 7;
        float* sprow = Sp + r * SSTRIDE;
        const unsigned char* mrow = MASK + (size_t)(b * L_ + q0 + r) * L_;

        // additive mask: -1e9 where mask byte == 0
        for (int i = 0; i < 64; ++i) {
            const int c4 = (p + 8 * i) * 4;
            const unsigned int mw = *(const unsigned int*)(mrow + c4);
            v4f s = *(v4f*)(sprow + c4);
            s[0] = (mw & 0x000000FFu) ? s[0] : s[0] - 1e9f;
            s[1] = (mw & 0x0000FF00u) ? s[1] : s[1] - 1e9f;
            s[2] = (mw & 0x00FF0000u) ? s[2] : s[2] - 1e9f;
            s[3] = (mw & 0xFF000000u) ? s[3] : s[3] - 1e9f;
            *(v4f*)(sprow + c4) = s;
        }
        float mx = -3.4e38f;
        for (int i = 0; i < 64; ++i) {
            const v4f s = *(const v4f*)(sprow + (p + 8 * i) * 4);
            mx = fmaxf(mx, fmaxf(fmaxf(s[0], s[1]), fmaxf(s[2], s[3])));
        }
        red[r * 8 + p] = mx;
        __syncthreads();
        if (p == 0) {
            float m2 = red[r * 8];
            #pragma unroll
            for (int j = 1; j < 8; ++j) m2 = fmaxf(m2, red[r * 8 + j]);
            rowmax[r] = m2;
        }
        __syncthreads();
        const float rmx = rowmax[r];
        float sm = 0.f;
        for (int i = 0; i < 64; ++i) {
            const v4f s = *(const v4f*)(sprow + (p + 8 * i) * 4);
            sm += __expf(s[0] - rmx) + __expf(s[1] - rmx)
                + __expf(s[2] - rmx) + __expf(s[3] - rmx);
        }
        red[r * 8 + p] = sm;
        __syncthreads();
        if (p == 0) {
            float s2 = 0.f;
            #pragma unroll
            for (int j = 0; j < 8; ++j) s2 += red[r * 8 + j];
            rowinv[r] = 1.0f / s2;
        }
        __syncthreads();
    }

    // ---- Phase 3: normalize, write attn (2KB/instr coalesced), keep P -----
    {
        float* arow0 = ATTN + ((size_t)(b * L_ + q0)) * L_;
        for (int i4 = tid; i4 < 16 * (L_ / 4); i4 += 128) {
            const int r  = i4 >> 9;           // / (2048/4)
            const int c  = (i4 & 511) * 4;
            const float rmx = rowmax[r];
            const float rin = rowinv[r];
            v4f s = *(v4f*)(Sp + r * SSTRIDE + c);
            s[0] = __expf(s[0] - rmx) * rin;
            s[1] = __expf(s[1] - rmx) * rin;
            s[2] = __expf(s[2] - rmx) * rin;
            s[3] = __expf(s[3] - rmx) * rin;
            *(v4f*)(Sp + r * SSTRIDE + c) = s;
            *(v4f*)(arow0 + (size_t)r * L_ + c) = s;
        }
        __syncthreads();
    }

    // ---------- Phase 4: OUT = P @ V, V staged transposed in LDS -----------
    {
        const int nA = wv * 32;              // this wave: out cols [nA, nA+32)
        v8f acc0 = {}, acc1 = {};

        stage_v_tile(Vt, vb, 0, tid);
        __syncthreads();

        int buf = 0;
        for (int kc = 0; kc < L_ / 32; ++kc) {
            if (kc + 1 < L_ / 32)
                stage_v_tile(Vt + (buf ^ 1) * VT_BUF, vb, (kc + 1) * 32, tid);

            // A fragment: normalized P from Sp panel (aligned b128 LDS loads)
            const float* sprow = Sp + m * SSTRIDE + kc * 32 + lb;
            const v4f p0 = *(const v4f*)(sprow);
            const v4f p1 = *(const v4f*)(sprow + 4);
            const v4f p2 = *(const v4f*)(sprow + 16);
            const v4f p3 = *(const v4f*)(sprow + 20);
            v16h pa;
            #pragma unroll
            for (int j = 0; j < 4; ++j) {
                pa[j]      = (_Float16)p0[j];
                pa[4 + j]  = (_Float16)p1[j];
                pa[8 + j]  = (_Float16)p2[j];
                pa[12 + j] = (_Float16)p3[j];
            }

            // B fragments: contiguous 8-half runs from transposed V tile
            const _Float16* vt  = Vt + buf * VT_BUF;
            const _Float16* c0p = vt + (nA + m) * VT_STRIDE + lb;
            const _Float16* c1p = vt + (nA + 16 + m) * VT_STRIDE + lb;
            const v8h b0a = *(const v8h*)(c0p);
            const v8h b0b = *(const v8h*)(c0p + 16);
            const v8h b1a = *(const v8h*)(c1p);
            const v8h b1b = *(const v8h*)(c1p + 16);
            const v16h bf0 = __builtin_shufflevector(b0a, b0b,
                0,1,2,3,4,5,6,7,8,9,10,11,12,13,14,15);
            const v16h bf1 = __builtin_shufflevector(b1a, b1b,
                0,1,2,3,4,5,6,7,8,9,10,11,12,13,14,15);

            acc0 = __builtin_amdgcn_wmma_f32_16x16x32_f16(
                       false, pa, false, bf0, (short)0, acc0, false, false);
            acc1 = __builtin_amdgcn_wmma_f32_16x16x32_f16(
                       false, pa, false, bf1, (short)0, acc1, false, false);

            __syncthreads();
            buf ^= 1;
        }

        float* orow0 = OUT + ((size_t)(b * L_ + q0)) * D_;
        #pragma unroll
        for (int vv = 0; vv < 8; ++vv) {
            const int mm = hi * 8 + vv;
            orow0[(size_t)mm * D_ + nA + m]      = acc0[vv];
            orow0[(size_t)mm * D_ + nA + 16 + m] = acc1[vv];
        }
    }
}

extern "C" void kernel_launch(void* const* d_in, const int* in_sizes, int n_in,
                              void* d_out, int out_size, void* d_ws, size_t ws_size,
                              hipStream_t stream) {
    (void)in_sizes; (void)n_in; (void)out_size; (void)d_ws; (void)ws_size;
    const float*         q    = (const float*)d_in[0];
    const float*         k    = (const float*)d_in[1];
    const float*         v    = (const float*)d_in[2];
    const unsigned char* mask = (const unsigned char*)d_in[3];

    float* out  = (float*)d_out;                       // [B, L, D]
    float* attn = out + (size_t)B_ * L_ * D_;          // [B, L, L]

    const size_t smem = (size_t)(16 * SSTRIDE + 128 + 16 + 16) * sizeof(float)
                      + (size_t)2 * VT_BUF * sizeof(_Float16);   // ~149 KB
    hipFuncSetAttribute((const void*)attn_fused_kernel,
                        hipFuncAttributeMaxDynamicSharedMemorySize, (int)smem);

    dim3 grid(B_ * (L_ / 16));   // 8192 workgroups
    dim3 block(128);             // 4 wave32
    attn_fused_kernel<<<grid, block, smem, stream>>>(q, k, v, mask, out, attn);
}
